// Stage2Loss_66992899883599
// MI455X (gfx1250) — matrix operand order repaired and checked by reading, not compiled, not run on hardware
//
#include <hip/hip_runtime.h>
#include <math.h>

// Problem constants (from reference setup_inputs)
#define B_SZ   512      // batch
#define H_SZ   256      // hardware qubits (K dim)
#define L_SZ   200      // logical qubits
#define E_SZ   512      // edges per pair set
#define QSH    32       // MLP hidden
#define LPAD   208      // 200 padded to 13*16
#define MT     13       // M tiles (208/16)
#define PST    264      // LDS row stride in bf16 elems (mult of 8 -> 16B aligned rows)
#define KT     8        // 256 / 32 k-steps per WMMA tile

typedef __attribute__((ext_vector_type(16))) __bf16 v16bf;
typedef __attribute__((ext_vector_type(8)))  __bf16 v8bf;
typedef __attribute__((ext_vector_type(8)))  float  v8f;

// ---- workspace layout (floats unless noted) ----
#define WSF_INVMAX  0
#define WSF_INVWSUM 1
#define WSF_Q       16                          // 256 floats: quality scores
#define WSF_CS      512                         // 208*208 edge-count matrix (surr)
#define WSF_CX      (512 + LPAD*LPAD)           // 208*208 edge-count matrix (sep)
#define WSF_PART    (512 + 2*LPAD*LPAD)         // 512*3 per-batch partials
#define WS_DTE_BYTE (((WSF_PART) + 3*B_SZ) * 4) // bf16 D_err^T (256x256)
#define WS_DTH_BYTE (WS_DTE_BYTE + H_SZ*H_SZ*2) // bf16 D_hw_norm^T (256x256)

#define SMEM_BYTES (LPAD*PST*2*2 + (H_SZ + LPAD + 256)*4)

// load 16 bf16 as two 16-byte chunks separated by `gap` elements
__device__ __forceinline__ v16bf ldpair(const __bf16* p, int gap) {
  v8bf a = *(const v8bf*)p;
  v8bf b = *(const v8bf*)(p + gap);
  return __builtin_shufflevector(a, b, 0,1,2,3,4,5,6,7,8,9,10,11,12,13,14,15);
}

// ---------------- kernel 1: scalars, MLP q, edge-count matrices ----------------
__global__ __launch_bounds__(1024)
void k_prep(const float* __restrict__ d_hw, const float* __restrict__ qimp,
            const float* __restrict__ feat, const float* __restrict__ W1,
            const float* __restrict__ b1,   const float* __restrict__ W2,
            const float* __restrict__ b2,
            const int* __restrict__ pairs_s, const int* __restrict__ pairs_x,
            float* __restrict__ wsf) {
  __shared__ float sred[1024];
  const int tid = threadIdx.x;
  float* Cs = wsf + WSF_CS;
  float* Cx = wsf + WSF_CX;
  for (int i = tid; i < LPAD*LPAD; i += 1024) { Cs[i] = 0.f; Cx[i] = 0.f; }

  // max(d_hw)
  float m = 0.f;
  for (int i = tid; i < H_SZ*H_SZ; i += 1024) m = fmaxf(m, d_hw[i]);
  sred[tid] = m; __syncthreads();
  for (int s = 512; s > 0; s >>= 1) { if (tid < s) sred[tid] = fmaxf(sred[tid], sred[tid+s]); __syncthreads(); }
  const float dmax = sred[0]; __syncthreads();

  // sum(qubit_importance)
  float w = 0.f;
  for (int i = tid; i < L_SZ; i += 1024) w += qimp[i];
  sred[tid] = w; __syncthreads();
  for (int s = 512; s > 0; s >>= 1) { if (tid < s) sred[tid] += sred[tid+s]; __syncthreads(); }
  const float wsum = sred[0];

  if (tid == 0) {
    wsf[WSF_INVMAX]  = 1.f / fmaxf(dmax, 1e-8f);
    wsf[WSF_INVWSUM] = 1.f / fmaxf(wsum, 1e-8f);
  }

  // quality MLP: q[h] = sigmoid(relu(feat@W1+b1)@W2+b2)
  if (tid < H_SZ) {
    float hv[QSH];
    #pragma unroll
    for (int j = 0; j < QSH; j++) {
      float s = b1[j];
      #pragma unroll
      for (int k = 0; k < 5; k++) s += feat[tid*5 + k] * W1[k*QSH + j];
      hv[j] = fmaxf(s, 0.f);
    }
    float o = b2[0];
    #pragma unroll
    for (int j = 0; j < QSH; j++) o += hv[j] * W2[j];
    wsf[WSF_Q + tid] = 1.f / (1.f + expf(-o));
  }
  __syncthreads();   // zeroing of C complete for all threads

  // build edge-count matrices (exact small-integer float sums; order independent)
  for (int e = tid; e < E_SZ; e += 1024) {
    atomicAdd(&Cs[pairs_s[2*e]*LPAD + pairs_s[2*e+1]], 1.f);
    atomicAdd(&Cx[pairs_x[2*e]*LPAD + pairs_x[2*e+1]], 1.f);
  }
}

// ---------------- kernel 2: bf16 transposed D matrices ----------------
__global__ __launch_bounds__(256)
void k_cvt(const float* __restrict__ d_err, const float* __restrict__ d_hw,
           const float* __restrict__ wsf, char* __restrict__ wsb) {
  __bf16* Dte = (__bf16*)(wsb + WS_DTE_BYTE);
  __bf16* Dth = (__bf16*)(wsb + WS_DTH_BYTE);
  const float invmax = wsf[WSF_INVMAX];
  const int o = blockIdx.x * 256 + threadIdx.x;   // 65536 threads total
  const int n = o >> 8, k = o & 255;
  Dte[o] = (__bf16)d_err[k*H_SZ + n];             // D^T[n][k] = D[k][n]
  Dth[o] = (__bf16)(d_hw[k*H_SZ + n] * invmax);
}

// ---------------- kernel 3: per-batch fused GEMM-GEMM-contract ----------------
__global__ __launch_bounds__(256)
void k_main(const float* __restrict__ P, const float* __restrict__ qimp,
            float* __restrict__ wsf, char* __restrict__ wsb) {
  extern __shared__ char smem[];
  __bf16* Plds = (__bf16*)smem;              // 208 x 264 bf16: P_log[b] (padded)
  __bf16* Tlds = Plds + LPAD*PST;            // 208 x 264 bf16: T = P @ D
  float*  qs   = (float*)(Tlds + LPAD*PST);  // 256
  float*  wl   = qs + H_SZ;                  // 208 (w_norm, zero-padded)
  float*  red  = wl + LPAD;                  // 256 reduction scratch

  const int b    = blockIdx.x;
  const int tid  = threadIdx.x;
  const int lane = tid & 31, wave = tid >> 5;
  const int hi   = lane >> 4, ln = lane & 15;

  const float invwsum = wsf[WSF_INVWSUM];
  for (int i = tid; i < H_SZ; i += 256) qs[i] = wsf[WSF_Q + i];
  for (int i = tid; i < LPAD; i += 256) wl[i] = (i < L_SZ) ? qimp[i]*invwsum : 0.f;
  for (int i = tid; i < (LPAD - L_SZ)*PST; i += 256) Plds[L_SZ*PST + i] = (__bf16)0.f;
  __syncthreads();

  // Load P_log[b] -> bf16 LDS; fuse node-loss accumulation (exact f32)
  float nodeacc = 0.f;
  const float* Pb = P + (size_t)b * H_SZ * H_SZ;
  for (int i = tid; i < L_SZ*H_SZ; i += 256) {
    const int r = i >> 8, c = i & 255;
    const float v = Pb[r*H_SZ + c];
    Plds[r*PST + c] = (__bf16)v;
    nodeacc += v * qs[c] * wl[r];
  }
  __syncthreads();

  float costAcc[2];
  for (int cost = 0; cost < 2; cost++) {
    const __bf16* Dt = (const __bf16*)(wsb + (cost ? WS_DTH_BYTE : WS_DTE_BYTE));
    const float*  C  = wsf + (cost ? WSF_CX : WSF_CS);

    // ---- GEMM1: T[208x256] = Plds[208x256] @ D[256x256]  (B from transposed Dt) ----
    for (int t = wave; t < MT*16; t += 8) {
      const int mt = t >> 4, nt = t & 15;
      v8f acc = {0.f,0.f,0.f,0.f,0.f,0.f,0.f,0.f};
      const __bf16* Ab = Plds + (mt*16 + ln)*PST + (hi ? 8 : 0);   // A: K 0-7/16-23 | 8-15/24-31
      const __bf16* Bb = Dt   + (nt*16 + ln)*H_SZ + (hi ? 8 : 0);  // B: N=ln, K 0-7/16-23 | 8-15/24-31
      #pragma unroll
      for (int kt = 0; kt < KT; kt++) {
        v16bf A  = ldpair(Ab + kt*32, 16);
        v16bf Bv = ldpair(Bb + kt*32, 16);
        acc = __builtin_amdgcn_wmma_f32_16x16x32_bf16(false, A, false, Bv,
                                                      (short)0, acc, false, false);
      }
      #pragma unroll
      for (int r = 0; r < 8; r++)   // D layout: (M = r + 8*hi, N = ln)
        Tlds[(mt*16 + 8*hi + r)*PST + nt*16 + ln] = (__bf16)acc[r];
    }
    __syncthreads();

    // ---- GEMM2 + C-contraction: part = sum C ⊙ (T @ P^T) ----
    float part = 0.f;
    for (int t = wave; t < MT*MT; t += 8) {
      const int mt = t / MT, nt = t % MT;
      v8f acc = {0.f,0.f,0.f,0.f,0.f,0.f,0.f,0.f};
      const __bf16* Ab = Tlds + (mt*16 + ln)*PST + (hi ? 8 : 0);
      const __bf16* Bb = Plds + (nt*16 + ln)*PST + (hi ? 8 : 0);  // B[k][n] = P[n][k]
      #pragma unroll
      for (int kt = 0; kt < KT; kt++) {
        v16bf A  = ldpair(Ab + kt*32, 16);
        v16bf Bv = ldpair(Bb + kt*32, 16);
        acc = __builtin_amdgcn_wmma_f32_16x16x32_bf16(false, A, false, Bv,
                                                      (short)0, acc, false, false);
      }
      const float* Crow = C + (mt*16 + 8*hi)*LPAD + nt*16 + ln;
      #pragma unroll
      for (int r = 0; r < 8; r++) part += acc[r] * Crow[r*LPAD];
    }
    costAcc[cost] = part;
    __syncthreads();   // Tlds reuse safe for next cost
  }

  // deterministic per-block reduction -> partials[b][3]
  float vals[3] = {costAcc[0], costAcc[1], nodeacc};
  for (int v = 0; v < 3; v++) {
    red[tid] = vals[v]; __syncthreads();
    for (int s = 128; s > 0; s >>= 1) { if (tid < s) red[tid] += red[tid+s]; __syncthreads(); }
    if (tid == 0) wsf[WSF_PART + 3*b + v] = red[0];
    __syncthreads();
  }
}

// ---------------- kernel 4: deterministic final combine ----------------
__global__ __launch_bounds__(256)
void k_final(const float* __restrict__ wsf, float* __restrict__ out) {
  __shared__ float r0[256], r1[256], r2[256];
  const int tid = threadIdx.x;
  float a = 0.f, b = 0.f, c = 0.f;
  for (int i = tid; i < B_SZ; i += 256) {
    a += wsf[WSF_PART + 3*i + 0];
    b += wsf[WSF_PART + 3*i + 1];
    c += wsf[WSF_PART + 3*i + 2];
  }
  r0[tid] = a; r1[tid] = b; r2[tid] = c; __syncthreads();
  for (int s = 128; s > 0; s >>= 1) {
    if (tid < s) { r0[tid] += r0[tid+s]; r1[tid] += r1[tid+s]; r2[tid] += r2[tid+s]; }
    __syncthreads();
  }
  if (tid == 0) {
    const float inv_be = 1.f / (float)(B_SZ * E_SZ);
    const float l_surr = r0[0] * inv_be;
    const float l_sep  = -(r1[0] * inv_be);
    const float l_node = -(r2[0] / (float)B_SZ);
    const float total  = l_surr + 0.3f*l_node + 0.1f*l_sep;
    out[0] = total; out[1] = l_surr; out[2] = l_node; out[3] = l_sep;
  }
}

extern "C" void kernel_launch(void* const* d_in, const int* in_sizes, int n_in,
                              void* d_out, int out_size, void* d_ws, size_t ws_size,
                              hipStream_t stream) {
  (void)in_sizes; (void)n_in; (void)out_size; (void)ws_size;
  const float* P    = (const float*)d_in[0];
  const float* derr = (const float*)d_in[1];
  const float* dhw  = (const float*)d_in[2];
  const float* feat = (const float*)d_in[3];
  const float* qimp = (const float*)d_in[4];
  const float* W1   = (const float*)d_in[5];
  const float* b1   = (const float*)d_in[6];
  const float* W2   = (const float*)d_in[7];
  const float* b2   = (const float*)d_in[8];
  const int*   ps   = (const int*)d_in[9];
  const int*   px   = (const int*)d_in[10];
  float* wsf = (float*)d_ws;
  char*  wsb = (char*)d_ws;
  float* out = (float*)d_out;

  (void)hipFuncSetAttribute((const void*)k_main,
                            hipFuncAttributeMaxDynamicSharedMemorySize, SMEM_BYTES);

  k_prep <<<1,    1024, 0,          stream>>>(dhw, qimp, feat, W1, b1, W2, b2, ps, px, wsf);
  k_cvt  <<<H_SZ, 256,  0,          stream>>>(derr, dhw, wsf, wsb);
  k_main <<<B_SZ, 256,  SMEM_BYTES, stream>>>(P, qimp, wsf, wsb);
  k_final<<<1,    256,  0,          stream>>>(wsf, out);
}
